// WfcNN_44032004719010
// MI455X (gfx1250) — matrix operand order repaired and checked by reading, not compiled, not run on hardware
//
#include <hip/hip_runtime.h>
#include <hip/hip_bf16.h>

typedef __attribute__((ext_vector_type(16))) _Float16 v16h;
typedef __attribute__((ext_vector_type(8)))  _Float16 v8h;
typedef __attribute__((ext_vector_type(2)))  _Float16 v2h;
typedef __attribute__((ext_vector_type(8)))  float    v8f;

#define N_EXPERTS 30
#define WG 256           // 8 waves of 32
#define ROWS_PER_BLOCK 128
#define MAX_TILES 48     // sum(ceil(c_e/16)) <= 30 + 128/16 = 38

// Hardware tanh on CDNA5 (V_TANH_F32, single TRANS32 op).
__device__ __forceinline__ float fast_tanh(float x) {
#if __has_builtin(__builtin_amdgcn_tanhf)
    return __builtin_amdgcn_tanhf(x);
#elif __has_builtin(__builtin_amdgcn_tanh_f32)
    return __builtin_amdgcn_tanh_f32(x);
#else
    float e = __builtin_amdgcn_exp2f(x * 2.8853900817779268f);
    return 1.0f - 2.0f * __builtin_amdgcn_rcpf(e + 1.0f);
#endif
}

// Wave-local LDS ordering (same-wave RAW through LDS; waves have divergent
// tile trip counts so __syncthreads is not usable inside the tile loop).
__device__ __forceinline__ void wave_lds_fence() {
    asm volatile("s_wait_dscnt 0" ::: "memory");
}

// Stop the machine scheduler from merging live ranges of adjacent regions
// (keeps VGPR pressure < 256 so no s_set_vgpr_msb bank switching is needed).
__device__ __forceinline__ void sched_fence() {
#if __has_builtin(__builtin_amdgcn_sched_barrier)
    __builtin_amdgcn_sched_barrier(0);
#endif
}

__global__
__attribute__((amdgpu_flat_work_group_size(WG, WG)))
__attribute__((amdgpu_waves_per_eu(4)))   // VGPR budget 1024/4 = 256: no MSB churn
void wfc_moe_kernel(const float* __restrict__ X,
                    const int* __restrict__ Narr,
                    const int* __restrict__ Larr,
                    const int* __restrict__ Marr,
                    const float* __restrict__ W0, const float* __restrict__ b0,
                    const float* __restrict__ W1, const float* __restrict__ b1,
                    const float* __restrict__ Wa, const float* __restrict__ ba,
                    const float* __restrict__ Wb, const float* __restrict__ bb,
                    float* __restrict__ out,
                    int Btot, int nGroups)
{
    // ---- LDS (~188 KB, under CDNA5's 320 KB/WGP) ----
    __shared__ __align__(32) float    sW0[3 * 128];
    __shared__ float                  sB0[128];
    __shared__ float                  sB1[64];
    __shared__ __align__(32) _Float16 sW1t[64 * 128];           // [n][k] = W1[k][n], f16
    __shared__ __align__(32) _Float16 sWa[N_EXPERTS * 32 * 64]; // [e][j][k] = Wa[e][k][j], f16
    __shared__ float                  sBa[N_EXPERTS * 32];
    __shared__ float                  sWb[N_EXPERTS * 64];      // [e][j*2+o]
    __shared__ float                  sBb[N_EXPERTS * 2];
    __shared__ __align__(32) _Float16 sH2[ROWS_PER_BLOCK][64];  // block-wide h2, f16
    __shared__ float                  sGt[8][16][32];           // per-wave g tile
    // expert binning state
    __shared__ int sCnt[N_EXPERTS], sStart[N_EXPERTS], sFill[N_EXPERTS];
    __shared__ int sEid[ROWS_PER_BLOCK];
    __shared__ int sIdx[ROWS_PER_BLOCK];                        // binned order -> local sample
    __shared__ int sTileE[MAX_TILES], sTileBase[MAX_TILES], sTileCnt[MAX_TILES];
    __shared__ int sNTiles;

    const int tid  = threadIdx.x;
    const int lane = tid & 31;
    const int wv   = tid >> 5;

    // ---- cooperative weight preload (once per block) ----
    for (int i = tid; i < 3 * 128; i += WG) sW0[i] = W0[i];
    for (int i = tid; i < 128; i += WG)     sB0[i] = b0[i];
    for (int i = tid; i < 64; i += WG)      sB1[i] = b1[i];
    for (int i = tid; i < 64 * 128; i += WG) {
        int nn = i >> 7, k = i & 127;
        sW1t[i] = (_Float16)W1[k * 64 + nn];
    }
    for (int i = tid; i < N_EXPERTS * 32 * 64; i += WG) {
        int e = i >> 11, r = i & 2047, j = r >> 6, k = r & 63;
        sWa[i] = (_Float16)Wa[(e << 11) + k * 32 + j];
    }
    for (int i = tid; i < N_EXPERTS * 32; i += WG) sBa[i] = ba[i];
    for (int i = tid; i < N_EXPERTS * 64; i += WG) sWb[i] = Wb[i];
    for (int i = tid; i < N_EXPERTS * 2;  i += WG) sBb[i] = bb[i];
    __syncthreads();

    const int row = lane & 15;             // M/slot within a 16-row tile
    const int hlf = lane >> 4;             // 0/1 lane pairs per row
    const int kbA = (lane < 16) ? 0 : 16;  // A-fragment K base (contiguous k = 16h + s, matches B)
    const int kbB = (lane < 16) ? 0 : 16;  // B-fragment K base
    const int col = lane & 15;             // C/D column
    const int mb  = (lane < 16) ? 0 : 8;   // C/D row base

    for (int grp = blockIdx.x; grp < nGroups; grp += gridDim.x) {
        const int grpBase = grp * ROWS_PER_BLOCK;

        // ===== phase 1: bin 128 samples by expert id =====
        if (tid < N_EXPERTS) { sCnt[tid] = 0; sFill[tid] = 0; }
        __syncthreads();
        if (tid < ROWS_PER_BLOCK) {
            const int gs = grpBase + tid;
            const int g  = (gs < Btot) ? gs : (Btot - 1);
            const int nq = Narr[g], lq = Larr[g], mq = Marr[g];
            const int off = (nq == 1) ? 0 : (nq == 2) ? 1 : (nq == 3) ? 5 : 14;
            const int e = off + lq * lq + lq + mq;
            sEid[tid] = e;
            atomicAdd(&sCnt[e], 1);
            // prefetch next group's x row (global_prefetch_b8)
            const int nx = gs + (int)gridDim.x * ROWS_PER_BLOCK;
            if (nx < Btot) __builtin_prefetch(&X[nx * 3], 0, 3);
        }
        __syncthreads();
        if (tid == 0) {                       // serial: <=38 iterations
            int pos = 0, nt = 0;
            for (int e = 0; e < N_EXPERTS; ++e) {
                sStart[e] = pos;
                const int c = sCnt[e];
                for (int b = 0; b < c; b += 16) {
                    sTileE[nt] = e; sTileBase[nt] = pos + b;
                    sTileCnt[nt] = (c - b < 16) ? (c - b) : 16;
                    ++nt;
                }
                pos += c;
            }
            sNTiles = nt;
        }
        __syncthreads();
        if (tid < ROWS_PER_BLOCK) {
            const int e = sEid[tid];
            const int r = atomicAdd(&sFill[e], 1);
            sIdx[sStart[e] + r] = tid;
        }

        // ===== phase 2: trunk, one 16-row tile per wave, WMMA 128x64 =====
        {
            const int base16 = wv * 16;
            const int gRow = grpBase + base16 + row;
            const int gr   = (gRow < Btot) ? gRow : (Btot - 1);
            const float x0 = X[gr * 3 + 0];
            const float x1 = X[gr * 3 + 1];
            const float x2 = X[gr * 3 + 2];

            // layer 0 fused into f16 A-fragments (fence per chunk: bounded liveness)
            union AF { v16h v; v2h p[8]; };
            AF af[4];
            #pragma unroll
            for (int c = 0; c < 4; ++c) {
                #pragma unroll
                for (int p8 = 0; p8 < 8; ++p8) {
                    const int k = c * 32 + kbA + 2 * p8;
                    float v0 = fast_tanh(x0 * sW0[k]     + x1 * sW0[128 + k]     + x2 * sW0[256 + k]     + sB0[k]);
                    float v1 = fast_tanh(x0 * sW0[k + 1] + x1 * sW0[128 + k + 1] + x2 * sW0[256 + k + 1] + sB0[k + 1]);
                    auto pk = __builtin_amdgcn_cvt_pkrtz(v0, v1);
                    af[c].p[p8] = __builtin_bit_cast(v2h, pk);
                }
                sched_fence();
            }

            // 16x128 @ 128x64 : rolled over N-tiles to bound VGPR pressure
            #pragma unroll 1
            for (int t = 0; t < 4; ++t) {
                v8f acc = {};
                #pragma unroll
                for (int c = 0; c < 4; ++c) {
                    v16h bfr = *(const v16h*)&sW1t[(t * 16 + col) * 128 + c * 32 + kbB];
                    acc = __builtin_amdgcn_wmma_f32_16x16x32_f16(
                            false, af[c].v, false, bfr, (short)0, acc, false, false);
                }
                const float bias = sB1[t * 16 + col];
                #pragma unroll
                for (int r = 0; r < 8; ++r)
                    sH2[base16 + mb + r][t * 16 + col] = (_Float16)fast_tanh(acc[r] + bias);
            }
        }
        __syncthreads();   // scatter + sH2 complete, visible block-wide

        // ===== phase 3: binned expert tiles on WMMA =====
        const int nT = sNTiles;
        for (int ti = wv; ti < nT; ti += 8) {
            const int e    = sTileE[ti];
            const int base = sTileBase[ti];
            const int cnt  = sTileCnt[ti];
            const int sl   = (row < cnt) ? row : (cnt - 1);   // clamp padded slots
            const int ls   = sIdx[base + sl];                 // local sample index

            // gather A-fragments (16x64 f16, 2 chunks) from binned h2 rows
            union AH { v16h v; v8h h[2]; };
            AH a2[2];
            #pragma unroll
            for (int c = 0; c < 2; ++c) {
                a2[c].h[0] = *(const v8h*)&sH2[ls][c * 32 + kbA];
                a2[c].h[1] = *(const v8h*)&sH2[ls][c * 32 + kbA + 8];
            }
            // 16x64 @ 64x32 : 2 N-tiles x 2 K-chunks of WMMA
            #pragma unroll
            for (int tN = 0; tN < 2; ++tN) {
                v8f acc = {};
                #pragma unroll
                for (int c = 0; c < 2; ++c) {
                    v16h bfr = *(const v16h*)&sWa[e * 2048 + (tN * 16 + col) * 64 + c * 32 + kbB];
                    acc = __builtin_amdgcn_wmma_f32_16x16x32_f16(
                            false, a2[c].v, false, bfr, (short)0, acc, false, false);
                }
                const float bias = sBa[e * 32 + tN * 16 + col];
                #pragma unroll
                for (int r = 0; r < 8; ++r)
                    sGt[wv][mb + r][tN * 16 + col] = fast_tanh(acc[r] + bias);
            }
            wave_lds_fence();

            // head: psi = g . Wb[e] + bb[e], then normalize (lane hlf = component)
            const int o = hlf;
            float acc = sBb[e * 2 + o];
            #pragma unroll
            for (int j = 0; j < 32; ++j)
                acc += sGt[wv][row][j] * sWb[e * 64 + j * 2 + o];
            const float other = __shfl_xor(acc, 16, 32);
            const float s = acc * acc + other * other + 1e-6f;
            const float res = acc * __builtin_amdgcn_rsqf(s);
            if (row < cnt) {
                const int gs = grpBase + ls;
                if (gs < Btot) out[gs * 2 + o] = res;
            }
            wave_lds_fence();  // protect sGt before next tile's writes
        }
        __syncthreads();       // protect sH2/bins before next group
    }
}

extern "C" void kernel_launch(void* const* d_in, const int* in_sizes, int n_in,
                              void* d_out, int out_size, void* d_ws, size_t ws_size,
                              hipStream_t stream) {
    (void)n_in; (void)d_ws; (void)ws_size; (void)out_size;
    const float* X  = (const float*)d_in[0];
    const int*   Nq = (const int*)d_in[1];
    const int*   Lq = (const int*)d_in[2];
    const int*   Mq = (const int*)d_in[3];
    const float* W0 = (const float*)d_in[4];
    const float* b0 = (const float*)d_in[5];
    const float* W1 = (const float*)d_in[6];
    const float* b1 = (const float*)d_in[7];
    const float* Wa = (const float*)d_in[8];
    const float* ba = (const float*)d_in[9];
    const float* Wb = (const float*)d_in[10];
    const float* bb = (const float*)d_in[11];
    float* out = (float*)d_out;

    const int Btot = in_sizes[0] / 3;
    const int nGroups = (Btot + ROWS_PER_BLOCK - 1) / ROWS_PER_BLOCK;
    int grid = nGroups < 512 ? nGroups : 512;
    hipLaunchKernelGGL(wfc_moe_kernel, dim3(grid), dim3(WG), 0, stream,
                       X, Nq, Lq, Mq, W0, b0, W1, b1, Wa, ba, Wb, bb,
                       out, Btot, nGroups);
}
